// RewireNetNodeClassification_14259291423163
// MI455X (gfx1250) — matrix operand order repaired and checked by reading, not compiled, not run on hardware
//
#include <hip/hip_runtime.h>
#include <hip/hip_bf16.h>
#include <math.h>
#include <stdint.h>

// ---------------- problem constants ----------------
#define Nn   3000
#define Ff   2344
#define Kk   11
#define NG   14
#define Gg   400
#define NC   7
#define NP   3072   // padded N (24*128, 96*32)
#define FP   2432   // padded F (19*128, 76*32)

// ---------------- types ----------------
typedef __bf16 bf16;
typedef __attribute__((ext_vector_type(8)))  __bf16 bf16x8;
typedef __attribute__((ext_vector_type(16))) __bf16 v16bf;
typedef __attribute__((ext_vector_type(8)))  float  v8f;

union Frag { bf16x8 h[2]; v16bf v; };

// ---------------- workspace layout (bytes) ----------------
#define OFF_CN    0u
#define OFF_MISC  64u          // float[8]: [0]=sum of group means, [1..7]=class sums
#define OFF_CNT   128u         // int[8] class counts
#define OFF_LHAT  256u                                   // bf16 NP*NP
#define OFF_XT    (OFF_LHAT + (size_t)NP*NP*2)           // bf16 FP*NP  (x transposed)
#define OFF_YHAT  (OFF_XT   + (size_t)FP*NP*2)           // bf16 NP*FP
#define OFF_SQ    (OFF_YHAT + (size_t)NP*FP*2)           // float NP
#define OFF_D     (OFF_SQ   + (size_t)NP*4)              // float NP*NP

// ---------------- CDNA5 async-to-LDS helpers ----------------
// global_load_async_to_lds_b128 vdst(lds addr), vaddr(64b global), off
// Tracked with ASYNCcnt; in-order completion per wave.
__device__ __forceinline__ void async_ld_b128(const bf16* gptr, bf16* lptr) {
    uint32_t l = (uint32_t)(uintptr_t)lptr;   // low 32 bits of generic = LDS offset
    uint64_t g = (uint64_t)(uintptr_t)gptr;
    asm volatile("global_load_async_to_lds_b128 %0, %1, off"
                 :: "v"(l), "v"(g) : "memory");
}

#if defined(__has_builtin) && __has_builtin(__builtin_amdgcn_s_wait_asynccnt)
#define WAIT_ASYNC(n) __builtin_amdgcn_s_wait_asynccnt(n)
#else
#define WAIT_ASYNC(n) asm volatile("s_wait_asynccnt %0" :: "i"(n) : "memory")
#endif

// ======================================================================
// 0) prep: Cn = C/||C||, zero accumulators
// ======================================================================
__global__ void prep_kernel(const float* __restrict__ C, float* cn, float* misc, int* cnt) {
    if (threadIdx.x == 0) {
        float nrm = 0.f;
        for (int k = 0; k < Kk; ++k) nrm += C[k] * C[k];
        float inv = 1.f / fmaxf(sqrtf(nrm), 1e-12f);
        for (int k = 0; k < Kk; ++k) cn[k] = C[k] * inv;
        for (int i = 0; i < 8; ++i) { misc[i] = 0.f; cnt[i] = 0; }
    }
}

// ======================================================================
// 1) L_hat[n,m] = sum_k Dbasis[n,m,k]*Cn[k]  -> bf16, zero-padded to NP*NP
// ======================================================================
__global__ __launch_bounds__(256) void lhat_kernel(const float* __restrict__ Db,
                                                   const float* __restrict__ cn,
                                                   bf16* __restrict__ L) {
    __shared__ float sc[Kk];
    if (threadIdx.x < Kk) sc[threadIdx.x] = cn[threadIdx.x];
    __syncthreads();
    size_t idx = (size_t)blockIdx.x * blockDim.x + threadIdx.x;
    if (idx >= (size_t)NP * NP) return;
    int n = (int)(idx / NP), m = (int)(idx % NP);
    float s = 0.f;
    if (n < Nn && m < Nn) {
        const float* p = Db + ((size_t)n * Nn + m) * Kk;
        #pragma unroll
        for (int k = 0; k < Kk; ++k) s += p[k] * sc[k];
    }
    L[idx] = (bf16)s;
}

// ======================================================================
// 2) xT[f,n] = bf16(x[n,f]), zero-padded to FP*NP  (LDS-tiled transpose)
// ======================================================================
__global__ __launch_bounds__(256) void transpose_x_kernel(const float* __restrict__ x,
                                                          bf16* __restrict__ xT) {
    __shared__ float tile[32][33];
    int tx = threadIdx.x, ty = threadIdx.y;
    int fbase = blockIdx.x * 32;   // feature dim
    int nbase = blockIdx.y * 32;   // node dim
    #pragma unroll
    for (int i = 0; i < 4; ++i) {
        int r = nbase + ty + i * 8, c = fbase + tx;
        tile[ty + i * 8][tx] = (r < Nn && c < Ff) ? x[(size_t)r * Ff + c] : 0.f;
    }
    __syncthreads();
    #pragma unroll
    for (int i = 0; i < 4; ++i) {
        int fr = fbase + ty + i * 8, nc = nbase + tx;
        xT[(size_t)fr * NP + nc] = (bf16)tile[tx][ty + i * 8];
    }
}

// ======================================================================
// BF16 WMMA GEMM core: block 128x128, 8 waves of 64x32, BK=32.
// A: M x K row-major (lda), Bt: N x K row-major (ldb) [Bt = B transposed].
// Double-buffered LDS fed by global_load_async_to_lds_b128 (ASYNCcnt).
// Epi(row, col, accval) stores one output element.
// ======================================================================
#define BM 128
#define BN 128
#define BK 32
#define LDT 40   // LDS row stride in bf16 (80B, bank-spread padding)

template <typename Epi>
__device__ __forceinline__ void gemm_core(const bf16* __restrict__ A,
                                          const bf16* __restrict__ Bt,
                                          int lda, int ldb, int ksteps,
                                          int m0, int n0, Epi epi) {
    __shared__ bf16 As[2][BM * LDT];
    __shared__ bf16 Bs[2][BN * LDT];
    const int tid = threadIdx.x, lane = tid & 31, wave = tid >> 5;
    const int wm = wave >> 2, wn = wave & 3;
    const int hf = lane >> 4, l16 = lane & 15;
    const int lr = tid >> 2, lc = (tid & 3) * 8;   // 64 rows/iter, 8 bf16 per load

    // prologue: async-stage tile 0 into buffer 0 (4 x b128 per thread)
    {
        #pragma unroll
        for (int i = 0; i < 2; ++i) {
            int r = lr + i * 64;
            async_ld_b128(&A [(size_t)(m0 + r) * lda + lc], &As[0][r * LDT + lc]);
            async_ld_b128(&Bt[(size_t)(n0 + r) * ldb + lc], &Bs[0][r * LDT + lc]);
        }
    }

    v8f acc[4][2] = {};
    for (int kt = 0; kt < ksteps; ++kt) {
        const int cur = kt & 1;
        if (kt + 1 < ksteps) {
            const int k1 = (kt + 1) * BK;
            #pragma unroll
            for (int i = 0; i < 2; ++i) {
                int r = lr + i * 64;
                async_ld_b128(&A [(size_t)(m0 + r) * lda + k1 + lc], &As[cur ^ 1][r * LDT + lc]);
                async_ld_b128(&Bt[(size_t)(n0 + r) * ldb + k1 + lc], &Bs[cur ^ 1][r * LDT + lc]);
            }
            WAIT_ASYNC(4);   // in-order: tile kt's 4 loads done, only kt+1's remain
        } else {
            WAIT_ASYNC(0);
        }
        __syncthreads();     // all waves' async writes to buf[cur] visible

        Frag af[4], bfr[2];
        #pragma unroll
        for (int mi = 0; mi < 4; ++mi) {
            const bf16* p = &As[cur][(wm * 64 + mi * 16 + l16) * LDT];
            af[mi].h[0] = *(const bf16x8*)(p + hf * 8);
            af[mi].h[1] = *(const bf16x8*)(p + 16 + hf * 8);
        }
        #pragma unroll
        for (int ni = 0; ni < 2; ++ni) {
            const bf16* p = &Bs[cur][(wn * 32 + ni * 16 + l16) * LDT];
            bfr[ni].h[0] = *(const bf16x8*)(p + hf * 8);
            bfr[ni].h[1] = *(const bf16x8*)(p + 16 + hf * 8);
        }
        #pragma unroll
        for (int mi = 0; mi < 4; ++mi)
            #pragma unroll
            for (int ni = 0; ni < 2; ++ni)
                acc[mi][ni] = __builtin_amdgcn_wmma_f32_16x16x32_bf16(
                    false, af[mi].v, false, bfr[ni].v, (short)0, acc[mi][ni], false, false);
        __syncthreads();     // done reading buf[cur]; safe to overwrite next iter
    }

    #pragma unroll
    for (int mi = 0; mi < 4; ++mi)
        #pragma unroll
        for (int ni = 0; ni < 2; ++ni)
            #pragma unroll
            for (int v = 0; v < 8; ++v) {
                int row = m0 + wm * 64 + mi * 16 + hf * 8 + v;
                int col = n0 + wn * 32 + ni * 16 + l16;
                epi(row, col, acc[mi][ni][v]);
            }
}

// 3) y_hat = x - L_hat @ x   (A=L_hat bf16, Bt=xT bf16, out bf16 NP x FP)
__global__ __launch_bounds__(256) void gemm1_kernel(const bf16* __restrict__ A,
                                                    const bf16* __restrict__ Bt,
                                                    const float* __restrict__ x,
                                                    bf16* __restrict__ Y) {
    const int m0 = blockIdx.y * BM, n0 = blockIdx.x * BN;
    gemm_core(A, Bt, NP, NP, NP / BK, m0, n0,
        [=](int row, int col, float a) {
            float xv = (row < Nn && col < Ff) ? x[(size_t)row * Ff + col] : 0.f;
            Y[(size_t)row * FP + col] = (bf16)(xv - a);
        });
}

// 4) sq[i] = sum_f y_hat[i,f]^2
__global__ __launch_bounds__(256) void sqnorm_kernel(const bf16* __restrict__ Y,
                                                     float* __restrict__ sq) {
    int row = blockIdx.x;
    float s = 0.f;
    for (int f = threadIdx.x; f < FP; f += blockDim.x) {
        float v = (float)Y[(size_t)row * FP + f];
        s += v * v;
    }
    __shared__ float red[256];
    red[threadIdx.x] = s; __syncthreads();
    for (int off = 128; off; off >>= 1) {
        if (threadIdx.x < off) red[threadIdx.x] += red[threadIdx.x + off];
        __syncthreads();
    }
    if (threadIdx.x == 0) sq[row] = red[0];
}

// 5) d[i,j] = sqrt(max(sq[i]+sq[j]-2*(y_hat y_hat^T)[i,j], 0)) with d2<=0 -> 0
__global__ __launch_bounds__(256) void gemm2_kernel(const bf16* __restrict__ A,
                                                    const float* __restrict__ sq,
                                                    float* __restrict__ D) {
    const int m0 = blockIdx.y * BM, n0 = blockIdx.x * BN;
    gemm_core(A, A, FP, FP, FP / BK, m0, n0,
        [=](int row, int col, float a) {
            float d2 = sq[row] + sq[col] - 2.f * a;
            D[(size_t)row * NP + col] = (d2 > 0.f) ? sqrtf(d2) : 0.f;
        });
}

// 6) hl1 accumulation: sum over groups of mean(d[g,g])
__global__ __launch_bounds__(256) void hl1_kernel(const float* __restrict__ D,
                                                  const int* __restrict__ ng,
                                                  float* __restrict__ sumMeans) {
    const int* gi = ng + blockIdx.x * Gg;
    float s = 0.f;
    for (int p = threadIdx.x; p < Gg * Gg; p += blockDim.x) {
        int a = gi[p / Gg], b = gi[p % Gg];
        s += D[(size_t)a * NP + b];
    }
    __shared__ float red[256];
    red[threadIdx.x] = s; __syncthreads();
    for (int off = 128; off; off >>= 1) {
        if (threadIdx.x < off) red[threadIdx.x] += red[threadIdx.x + off];
        __syncthreads();
    }
    if (threadIdx.x == 0) atomicAdd(sumMeans, red[0] / (float)(Gg * Gg));
}

// 7) class counts
__global__ void cnt_kernel(const int* __restrict__ y, const unsigned char* __restrict__ mask,
                           int* __restrict__ cnt) {
    int i = blockIdx.x * blockDim.x + threadIdx.x;
    if (i < Nn && mask[i]) atomicAdd(&cnt[y[i]], 1);
}

// 8) class sums: s_c = sum_{a,b in class c} d[a,b]
__global__ __launch_bounds__(256) void csum_kernel(const float* __restrict__ D,
                                                   const int* __restrict__ y,
                                                   const unsigned char* __restrict__ mask,
                                                   float* __restrict__ csum) {
    int a = blockIdx.x;
    int c = y[a];
    float s = 0.f;
    if (mask[a]) {
        for (int b = threadIdx.x; b < Nn; b += blockDim.x)
            if (mask[b] && y[b] == c) s += D[(size_t)a * NP + b];
    }
    __shared__ float red[256];
    red[threadIdx.x] = s; __syncthreads();
    for (int off = 128; off; off >>= 1) {
        if (threadIdx.x < off) red[threadIdx.x] += red[threadIdx.x + off];
        __syncthreads();
    }
    if (threadIdx.x == 0 && mask[a]) atomicAdd(&csum[c], red[0]);
}

// 9) final scalar combine
__global__ void final_kernel(const float* __restrict__ C, const float* __restrict__ misc,
                             const int* __restrict__ cnt, float* __restrict__ out) {
    if (threadIdx.x != 0 || blockIdx.x != 0) return;
    float nrm = 0.f;
    for (int k = 0; k < Kk; ++k) nrm += C[k] * C[k];
    float inv = 1.f / fmaxf(sqrtf(nrm), 1e-12f);
    float l1 = 0.f, l2s = 0.f;
    for (int k = 0; k < Kk; ++k) {
        float cn = C[k] * inv;
        l1 += fabsf(cn); l2s += cn * cn;
    }
    float l2 = sqrtf(l2s);
    float dims = sqrtf((float)Kk);
    float sparsity = (dims - l1 / l2) / (dims - 1.f);
    float hl2 = 0.f;
    for (int i = 0; i < NC; ++i) {
        float c = (float)cnt[i];
        if (c > 0.f) hl2 += misc[1 + i] / fmaxf(c * c, 1.f);
    }
    float hl1 = -misc[0];
    float beta = (float)NG / (float)NC;
    out[0] = sparsity + hl2 + hl1 / beta;
}

// ======================================================================
extern "C" void kernel_launch(void* const* d_in, const int* in_sizes, int n_in,
                              void* d_out, int out_size, void* d_ws, size_t ws_size,
                              hipStream_t stream) {
    const float*         x    = (const float*)d_in[0];
    const float*         Db   = (const float*)d_in[1];
    const float*         C    = (const float*)d_in[2];
    const unsigned char* mask = (const unsigned char*)d_in[3];
    const int*           y    = (const int*)d_in[4];
    const int*           ng   = (const int*)d_in[5];
    float*               out  = (float*)d_out;

    char* ws = (char*)d_ws;
    float* cn    = (float*)(ws + OFF_CN);
    float* misc  = (float*)(ws + OFF_MISC);
    int*   cnt   = (int*)  (ws + OFF_CNT);
    bf16*  Lhat  = (bf16*) (ws + OFF_LHAT);
    bf16*  xT    = (bf16*) (ws + OFF_XT);
    bf16*  Yhat  = (bf16*) (ws + OFF_YHAT);
    float* sq    = (float*)(ws + OFF_SQ);
    float* D     = (float*)(ws + OFF_D);

    prep_kernel<<<1, 32, 0, stream>>>(C, cn, misc, cnt);
    lhat_kernel<<<(NP * NP) / 256, 256, 0, stream>>>(Db, cn, Lhat);
    transpose_x_kernel<<<dim3(FP / 32, NP / 32), dim3(32, 8), 0, stream>>>(x, xT);
    gemm1_kernel<<<dim3(FP / BN, NP / BM), 256, 0, stream>>>(Lhat, xT, x, Yhat);
    sqnorm_kernel<<<NP, 256, 0, stream>>>(Yhat, sq);
    gemm2_kernel<<<dim3(NP / BN, NP / BM), 256, 0, stream>>>(Yhat, sq, D);
    hl1_kernel<<<NG, 256, 0, stream>>>(D, ng, &misc[0]);
    cnt_kernel<<<(Nn + 255) / 256, 256, 0, stream>>>(y, mask, cnt);
    csum_kernel<<<Nn, 256, 0, stream>>>(D, y, mask, &misc[1]);
    final_kernel<<<1, 1, 0, stream>>>(C, misc, cnt, out);
}
